// My_GNN_NN_36155034698325
// MI455X (gfx1250) — matrix operand order repaired and checked by reading, not compiled, not run on hardware
//
#include <hip/hip_runtime.h>
#include <hip/hip_bf16.h>

typedef __attribute__((ext_vector_type(16))) _Float16 v16h;
typedef __attribute__((ext_vector_type(8)))  _Float16 v8h;
typedef __attribute__((ext_vector_type(8)))  float    v8f;

#define NEG_SLOPE 0.2f

__device__ __forceinline__ void atomicMaxF(float* a, float v) {
  if (v >= 0.0f) atomicMax((int*)a, __float_as_int(v));
  else           atomicMin((unsigned int*)a, __float_as_uint(v));
}

// ---------------------------------------------------------------- utilities
__global__ void fill_u32(unsigned* __restrict__ p, int n, unsigned v) {
  int i = blockIdx.x * blockDim.x + threadIdx.x;
  if (i < n) p[i] = v;
}

// Convert the two 128x128 f32 weight matrices to f16 AND repack into the
// v_wmma_f32_16x16x32_f16 B-fragment order this file uses (k = 16*half + j, as the A loads):
//   Bp[((ct*4 + kk)*32 + lane)*16 + j] = B[(kk*32 + (lane>>4)*16 + j)*128 + ct*16 + (lane&15)]
// so each lane's 16 fragment halves are one contiguous 32-byte chunk.
__global__ void cvt_pack_f16(const float* __restrict__ a, const float* __restrict__ b,
                             _Float16* __restrict__ ah, _Float16* __restrict__ bh) {
  int i = blockIdx.x * blockDim.x + threadIdx.x;
  if (i >= 32768) return;
  int p = (i < 16384) ? i : i - 16384;
  int j    = p & 15;
  int lane = (p >> 4) & 31;
  int kk   = (p >> 9) & 3;
  int ct   = (p >> 11) & 7;
  int src = (kk * 32 + (lane >> 4) * 16 + j) * 128 + ct * 16 + (lane & 15);
  if (i < 16384) ah[p] = (_Float16)a[src];
  else           bh[p] = (_Float16)b[src];
}

// ------------------------------------------------- conv1: transform (K=3) + att
// one wave32 per node; lane handles 4 of 128 channels; head = lane/8 (32 ch/head)
__global__ void conv1_node(const float* __restrict__ x, const float* __restrict__ W1,
                           const float* __restrict__ att_s, const float* __restrict__ att_d,
                           float* __restrict__ h_pre, float* __restrict__ a_s,
                           float* __restrict__ a_d, int N) {
  int wid = (blockIdx.x * blockDim.x + threadIdx.x) >> 5;
  int l = threadIdx.x & 31;
  if (wid >= N) return;
  float x0 = x[wid * 3 + 0], x1 = x[wid * 3 + 1], x2 = x[wid * 3 + 2];
  int c0 = l * 4;
  int head = l >> 3;
  float ps = 0.0f, pd = 0.0f;
  float* hp = h_pre + (size_t)wid * 128 + c0;
#pragma unroll
  for (int j = 0; j < 4; ++j) {
    int c = c0 + j;
    float h = x0 * W1[c] + x1 * W1[128 + c] + x2 * W1[256 + c];
    hp[j] = h;
    ps += h * att_s[c];   // (HEADS,HIDDEN) flat index == c
    pd += h * att_d[c];
  }
#pragma unroll
  for (int off = 1; off < 8; off <<= 1) {   // reduce within 8-lane head group
    ps += __shfl_xor(ps, off, 32);
    pd += __shfl_xor(pd, off, 32);
  }
  if ((l & 7) == 0) {
    a_s[wid * 4 + head] = ps;
    a_d[wid * 4 + head] = pd;
  }
}

// ------------------------------------------------- conv2 attention (H=1, C=128)
__global__ void att_node128(const float* __restrict__ h, const float* __restrict__ att_s,
                            const float* __restrict__ att_d, float* __restrict__ a_s,
                            float* __restrict__ a_d, int N) {
  int wid = (blockIdx.x * blockDim.x + threadIdx.x) >> 5;
  int l = threadIdx.x & 31;
  if (wid >= N) return;
  float ps = 0.0f, pd = 0.0f;
#pragma unroll
  for (int j = 0; j < 4; ++j) {
    int c = l * 4 + j;
    float v = h[(size_t)wid * 128 + c];
    ps += v * att_s[c];
    pd += v * att_d[c];
  }
#pragma unroll
  for (int off = 1; off < 32; off <<= 1) {
    ps += __shfl_xor(ps, off, 32);
    pd += __shfl_xor(pd, off, 32);
  }
  if (l == 0) { a_s[wid] = ps; a_d[wid] = pd; }
}

// ------------------------------------------------- edge pass 1: segment max
__global__ void edge_max(const int* __restrict__ ei, const float* __restrict__ a_s,
                         const float* __restrict__ a_d, float* __restrict__ m,
                         int E, int N, int H) {
  int e = blockIdx.x * blockDim.x + threadIdx.x;
  if (e >= E + N) return;
  int s, d;
  if (e < E) { s = ei[e]; d = ei[E + e]; } else { s = d = e - E; }
  for (int h = 0; h < H; ++h) {
    float v = a_s[s * H + h] + a_d[d * H + h];
    v = v > 0.0f ? v : NEG_SLOPE * v;
    atomicMaxF(&m[d * H + h], v);
  }
}

// ------------------------------------------------- edge pass 2: denom sum
__global__ void edge_sum(const int* __restrict__ ei, const float* __restrict__ a_s,
                         const float* __restrict__ a_d, const float* __restrict__ m,
                         float* __restrict__ denom, int E, int N, int H) {
  int e = blockIdx.x * blockDim.x + threadIdx.x;
  if (e >= E + N) return;
  int s, d;
  if (e < E) { s = ei[e]; d = ei[E + e]; } else { s = d = e - E; }
  for (int h = 0; h < H; ++h) {
    float v = a_s[s * H + h] + a_d[d * H + h];
    v = v > 0.0f ? v : NEG_SLOPE * v;
    atomicAdd(&denom[d * H + h], __expf(v - m[d * H + h]));
  }
}

// ------------------------------------------------- edge pass 3: weighted scatter
// one wave per edge, 4 channels per lane
__global__ void edge_aggr(const int* __restrict__ ei, const float* __restrict__ h_pre,
                          const float* __restrict__ a_s, const float* __restrict__ a_d,
                          const float* __restrict__ m, const float* __restrict__ denom,
                          float* __restrict__ out, int E, int N, int H) {
  long long gid = (long long)blockIdx.x * blockDim.x + threadIdx.x;
  int e = (int)(gid >> 5);
  int l = threadIdx.x & 31;
  if (e >= E + N) return;
  int s, d;
  if (e < E) { s = ei[e]; d = ei[E + e]; } else { s = d = e - E; }
  int head = (l * H) >> 5;                       // H=4 -> l/8, H=1 -> 0
  const float* hs = h_pre + (size_t)s * 128 + l * 4;
  __builtin_prefetch(hs, 0, 0);                  // global_prefetch_b8
  float ev = a_s[s * H + head] + a_d[d * H + head];
  ev = ev > 0.0f ? ev : NEG_SLOPE * ev;
  float alpha = __expf(ev - m[d * H + head]) / (denom[d * H + head] + 1e-16f);
  float* od = out + (size_t)d * 128 + l * 4;
#pragma unroll
  for (int j = 0; j < 4; ++j) atomicAdd(od + j, hs[j] * alpha);
}

// ------------------------------------------------- bias + relu + f16 convert
__global__ void bias_relu_f16(const float* __restrict__ in, const float* __restrict__ bias,
                              _Float16* __restrict__ outh, int n) {
  int i = blockIdx.x * blockDim.x + threadIdx.x;
  if (i < n) {
    float v = in[i] + bias[i & 127];
    outh[i] = (_Float16)fmaxf(v, 0.0f);
  }
}

// ------------------------------------------------- WMMA GEMM: C[M,128]=A[M,128]@Bp
// Bp is pre-packed in fragment order. One wave computes 16 rows x 32 cols
// (2 col tiles, 8 v_wmma, A fragment reused). 8 waves/block => 32 rows x 128 cols.
__global__ void __launch_bounds__(256)
gat_gemm_128(const _Float16* __restrict__ A, const _Float16* __restrict__ Bp,
             const float* __restrict__ bias, float* __restrict__ Cf,
             _Float16* __restrict__ Ch, int M, int relu) {
  const int lane = threadIdx.x & 31;
  const int w    = threadIdx.x >> 5;           // 0..7
  const int half = lane >> 4;
  const int l    = lane & 15;
  const int rowTile = blockIdx.x * 2 + (w >> 2);
  const int ct0     = (w & 3) * 2;             // this wave's col tiles: ct0, ct0+1
  const int rowBase = rowTile * 16;
  int arow = rowBase + l;
  if (arow >= M) arow = M - 1;                 // clamp: EXEC stays all-ones for WMMA
  const _Float16* Abase = A + (size_t)arow * 128;
  v8f acc0 = {}, acc1 = {};
#pragma unroll
  for (int kk = 0; kk < 4; ++kk) {
    const int kb = kk * 32;
    // A 16x32 fragment: one contiguous 32B run per lane, k = 16*half + s (matches Bp)
    v8h alo = *(const v8h*)(Abase + kb + half * 16);
    v8h ahi = *(const v8h*)(Abase + kb + half * 16 + 8);
    v16h a = __builtin_shufflevector(alo, ahi, 0, 1, 2, 3, 4, 5, 6, 7,
                                               8, 9, 10, 11, 12, 13, 14, 15);
    // B fragments: one contiguous 32B chunk per lane (pre-packed)
    v16h b0 = *(const v16h*)(Bp + (size_t)(((ct0 + 0) * 4 + kk) * 32 + lane) * 16);
    v16h b1 = *(const v16h*)(Bp + (size_t)(((ct0 + 1) * 4 + kk) * 32 + lane) * 16);
    acc0 = __builtin_amdgcn_wmma_f32_16x16x32_f16(false, a, false, b0,
                                                  (short)0, acc0, false, false);
    acc1 = __builtin_amdgcn_wmma_f32_16x16x32_f16(false, a, false, b1,
                                                  (short)0, acc1, false, false);
  }
#pragma unroll
  for (int t = 0; t < 2; ++t) {
    const v8f acc = t ? acc1 : acc0;
    const int n = (ct0 + t) * 16 + l;
    const float bn = bias ? bias[n] : 0.0f;
#pragma unroll
    for (int r = 0; r < 8; ++r) {              // C: VGPR r -> M = 8*half + r, N = l
      int mrow = rowBase + half * 8 + r;
      if (mrow < M) {
        float v = acc[r] + bn;
        if (relu) v = fmaxf(v, 0.0f);
        if (Cf) Cf[(size_t)mrow * 128 + n] = v;
        if (Ch) Ch[(size_t)mrow * 128 + n] = (_Float16)v;
      }
    }
  }
}

// ------------------------------------------------- lin2: 128 -> 2
__global__ void lin2_node(const _Float16* __restrict__ h, const float* __restrict__ W,
                          const float* __restrict__ b, float* __restrict__ out, int N) {
  int wid = (blockIdx.x * blockDim.x + threadIdx.x) >> 5;
  int l = threadIdx.x & 31;
  if (wid >= N) return;
  float p0 = 0.0f, p1 = 0.0f;
#pragma unroll
  for (int j = 0; j < 4; ++j) {
    int c = l * 4 + j;
    float v = (float)h[(size_t)wid * 128 + c];
    p0 += v * W[c * 2 + 0];
    p1 += v * W[c * 2 + 1];
  }
#pragma unroll
  for (int off = 1; off < 32; off <<= 1) {
    p0 += __shfl_xor(p0, off, 32);
    p1 += __shfl_xor(p1, off, 32);
  }
  if (l == 0) {
    out[wid * 2 + 0] = p0 + b[0];
    out[wid * 2 + 1] = p1 + b[1];
  }
}

// ================================================================ host
extern "C" void kernel_launch(void* const* d_in, const int* in_sizes, int n_in,
                              void* d_out, int out_size, void* d_ws, size_t ws_size,
                              hipStream_t stream) {
  const float* x   = (const float*)d_in[0];
  const int*   ei  = (const int*)d_in[1];
  const float* W1  = (const float*)d_in[2];
  const float* as1 = (const float*)d_in[3];
  const float* ad1 = (const float*)d_in[4];
  const float* b1  = (const float*)d_in[5];
  const float* W2  = (const float*)d_in[6];
  const float* as2 = (const float*)d_in[7];
  const float* ad2 = (const float*)d_in[8];
  const float* b2  = (const float*)d_in[9];
  const float* l1W = (const float*)d_in[10];
  const float* l1b = (const float*)d_in[11];
  const float* l2W = (const float*)d_in[12];
  const float* l2b = (const float*)d_in[13];
  float* out = (float*)d_out;

  const int N = in_sizes[0] / 3;
  const int E = in_sizes[1] / 2;
  const int tot = E + N;

  // ---- workspace layout (regions reused across phases)
  char* ws = (char*)d_ws;
  size_t off = 0;
  auto alloc = [&](size_t bytes) { size_t r = off; off = (off + bytes + 255) & ~(size_t)255; return r; };
  float*    h_pre = (float*)(ws + alloc((size_t)N * 128 * 4));    // h1_pre then h2_pre
  float*    agg   = (float*)(ws + alloc((size_t)N * 128 * 4));    // out1 then out2
  _Float16* hhA   = (_Float16*)(ws + alloc((size_t)N * 128 * 2)); // relu1(f16) then lin1 out(f16)
  _Float16* hhB   = (_Float16*)(ws + alloc((size_t)N * 128 * 2)); // relu2(f16)
  float*    a_s   = (float*)(ws + alloc((size_t)N * 4 * 4));
  float*    a_d   = (float*)(ws + alloc((size_t)N * 4 * 4));
  float*    mbuf  = (float*)(ws + alloc((size_t)N * 4 * 4));
  float*    dbuf  = (float*)(ws + alloc((size_t)N * 4 * 4));
  _Float16* W2p   = (_Float16*)(ws + alloc(128 * 128 * 2));       // pre-packed B fragments
  _Float16* l1Wp  = (_Float16*)(ws + alloc(128 * 128 * 2));

  const int T = 256;
  auto gr = [&](long long n) { return (unsigned)((n + T - 1) / T); };

  // weights -> f16, repacked into WMMA B-fragment order (deterministic each call)
  cvt_pack_f16<<<gr(32768), T, 0, stream>>>(W2, l1W, W2p, l1Wp);

  // ================= conv1 (3 -> 32 x 4 heads, concat)
  fill_u32<<<gr((long long)N * 128), T, 0, stream>>>((unsigned*)agg, N * 128, 0u);
  fill_u32<<<gr(N * 4), T, 0, stream>>>((unsigned*)mbuf, N * 4, 0xFF800000u); // -inf
  fill_u32<<<gr(N * 4), T, 0, stream>>>((unsigned*)dbuf, N * 4, 0u);
  conv1_node<<<gr((long long)N * 32), T, 0, stream>>>(x, W1, as1, ad1, h_pre, a_s, a_d, N);
  edge_max <<<gr(tot), T, 0, stream>>>(ei, a_s, a_d, mbuf, E, N, 4);
  edge_sum <<<gr(tot), T, 0, stream>>>(ei, a_s, a_d, mbuf, dbuf, E, N, 4);
  edge_aggr<<<gr((long long)tot * 32), T, 0, stream>>>(ei, h_pre, a_s, a_d, mbuf, dbuf, agg, E, N, 4);
  bias_relu_f16<<<gr((long long)N * 128), T, 0, stream>>>(agg, b1, hhA, N * 128);

  // ================= conv2 transform: h2_pre = relu1 @ W2   (WMMA)
  gat_gemm_128<<<(N + 31) / 32, T, 0, stream>>>(hhA, W2p, nullptr, h_pre, nullptr, N, 0);
  att_node128<<<gr((long long)N * 32), T, 0, stream>>>(h_pre, as2, ad2, a_s, a_d, N);

  // ================= conv2 softmax + aggregation (H=1)
  fill_u32<<<gr((long long)N * 128), T, 0, stream>>>((unsigned*)agg, N * 128, 0u);
  fill_u32<<<gr(N), T, 0, stream>>>((unsigned*)mbuf, N, 0xFF800000u);
  fill_u32<<<gr(N), T, 0, stream>>>((unsigned*)dbuf, N, 0u);
  edge_max <<<gr(tot), T, 0, stream>>>(ei, a_s, a_d, mbuf, E, N, 1);
  edge_sum <<<gr(tot), T, 0, stream>>>(ei, a_s, a_d, mbuf, dbuf, E, N, 1);
  edge_aggr<<<gr((long long)tot * 32), T, 0, stream>>>(ei, h_pre, a_s, a_d, mbuf, dbuf, agg, E, N, 1);
  bias_relu_f16<<<gr((long long)N * 128), T, 0, stream>>>(agg, b2, hhB, N * 128);

  // ================= lin1 (fused bias+relu, f16 out)   (WMMA)
  gat_gemm_128<<<(N + 31) / 32, T, 0, stream>>>(hhB, l1Wp, l1b, nullptr, hhA, N, 1);

  // ================= lin2 (128 -> 2)
  lin2_node<<<gr((long long)N * 32), T, 0, stream>>>(hhA, l2W, l2b, out, N);
}
